// JointNetwork_67439576482058
// MI455X (gfx1250) — matrix operand order repaired and checked by reading, not compiled, not run on hardware
//
#include <hip/hip_runtime.h>
#include <stdint.h>

// ---------------------------------------------------------------- types
typedef __attribute__((ext_vector_type(16))) __bf16 v16bf;
typedef __attribute__((ext_vector_type(8)))  float  v8f;
typedef int gli4 __attribute__((vector_size(16)));   // pointee type of async-LDS builtin

#define DEV static __device__ __forceinline__

// sizes (fixed by the reference)
#define BB 8
#define TT 256
#define UU 64
#define DD 512
#define JJ 640
#define VV 1024

// ---------------------------------------------------------------- async-to-LDS support
#if __has_builtin(__builtin_amdgcn_global_load_async_to_lds_b128)
#define HAVE_ASYNC_LDS 1
#else
#define HAVE_ASYNC_LDS 0
#endif

// AS1 pointer value == 64-bit flat VA; AS3 pointer value == low 32 bits of the
// flat VA (flat LDS address = aperture_hi || lds_offset, ISA 10.2). Build the
// address-space pointers via int->pointer casts (always legal, no addrspacecast).
// The builtin's params are pointers to 16B int vectors (per hipcc diagnostic).
#define AS1CAST(p) ((__attribute__((address_space(1))) gli4*)(uintptr_t)(p))
#define AS3CAST(p) ((__attribute__((address_space(3))) gli4*)(uint32_t)(uintptr_t)(p))

DEV void wait_async0() {
#if __has_builtin(__builtin_amdgcn_s_wait_asynccnt)
  __builtin_amdgcn_s_wait_asynccnt(0);
#else
  asm volatile("s_wait_asynccnt 0" ::: "memory");
#endif
}

// ---------------------------------------------------------------- helpers
DEV uint16_t f2bf(float x) {                 // fp32 -> bf16, round-to-nearest-even
  uint32_t u = __float_as_uint(x);
  u += 0x7FFFu + ((u >> 16) & 1u);
  return (uint16_t)(u >> 16);
}
DEV uint32_t pack2(float a, float b) {
  return (uint32_t)f2bf(a) | ((uint32_t)f2bf(b) << 16);
}
DEV float fast_tanh(float x) {
#if __has_builtin(__builtin_amdgcn_tanhf)
  return __builtin_amdgcn_tanhf(x);          // gfx1250 v_tanh_f32 (trans pipe)
#else
  return tanhf(x);
#endif
}

union Frag { uint4 q[2]; v16bf v; };         // 16 bf16 = 8 VGPRs per lane

DEV v8f wmma_bf16(const Frag& a, const Frag& b, v8f c) {
  // D = A(16x32 bf16) * B(32x16 bf16) + C(16x16 f32)
  return __builtin_amdgcn_wmma_f32_16x16x32_bf16(
      /*neg_a=*/false, a.v, /*neg_b=*/false, b.v,
      /*c_mod=*/(short)0, c, /*reuse_a=*/false, /*reuse_b=*/false);
}

// ---------------------------------------------------------------- stage 0: conversions
__global__ __launch_bounds__(256) void cvt_pairs(const float* __restrict__ s,
                                                 uint32_t* __restrict__ d, int np) {
  int i = blockIdx.x * 256 + threadIdx.x;
  if (i < np) {
    float2 v = ((const float2*)s)[i];
    d[i] = pack2(v.x, v.y);
  }
}

// src[K][N] f32 (row-major)  ->  dst[N][K] bf16 (row-major, i.e. transposed)
__global__ __launch_bounds__(256) void cvtT(const float* __restrict__ s,
                                            uint16_t* __restrict__ d, int K, int N) {
  int i = blockIdx.x * 256 + threadIdx.x;
  if (i < K * N) {
    int n = i / K;
    int k = i - n * K;
    d[i] = f2bf(s[(int64_t)k * N + n]);
  }
}

// ---------------------------------------------------------------- stage 1: projections
// P[M][640] = Abf[M][512](bf16) @ WT[640][512]^T + bias    (f32 out)
#define P_MT 64
#define P_NT 128
#define SA1 40
#define SB1 40

__global__ __launch_bounds__(256) void proj_gemm(const uint16_t* __restrict__ Abf,
                                                 const uint16_t* __restrict__ WT,
                                                 const float* __restrict__ bias,
                                                 float* __restrict__ P) {
  __shared__ uint16_t sA[2][P_MT * SA1];
  __shared__ uint16_t sB[2][P_NT * SB1];
  const int tid = threadIdx.x;
  const int m0 = blockIdx.x * P_MT;
  const int n0 = blockIdx.y * P_NT;

  const int lane = tid & 31, w = tid >> 5;
  const int wm = w & 1, wn = w >> 1;          // 2 M-waves x 4 N-waves
  const int ln = lane & 15, h = lane >> 4;

  const int am = tid >> 2, ap = tid & 3;      // A stage: 64 rows x 4x16B
  const int bn = tid >> 1, bp = tid & 1;      // B stage: 128 rows x 2x32B

#if HAVE_ASYNC_LDS
  auto stage = [&](int buf, int k) {          // DMA straight into LDS (ASYNCcnt)
    const uint16_t* ga = Abf + (int64_t)(m0 + am) * DD + k + ap * 8;
    uint16_t* la = &sA[buf][am * SA1 + ap * 8];
    __builtin_amdgcn_global_load_async_to_lds_b128(AS1CAST(ga), AS3CAST(la), 0, 0);
    const uint16_t* gb = WT + (int64_t)(n0 + bn) * DD + k + bp * 16;
    uint16_t* lb = &sB[buf][bn * SB1 + bp * 16];
    __builtin_amdgcn_global_load_async_to_lds_b128(AS1CAST(gb), AS3CAST(lb), 0, 0);
    __builtin_amdgcn_global_load_async_to_lds_b128(AS1CAST(gb), AS3CAST(lb), 16, 0);
  };
  stage(0, 0);
  wait_async0();
#else
  uint4 pa, pb0, pb1;
  auto gA = [&](int k) { pa = *(const uint4*)(Abf + (int64_t)(m0 + am) * DD + k + ap * 8); };
  auto gB = [&](int k) {
    const uint4* s = (const uint4*)(WT + (int64_t)(n0 + bn) * DD + k + bp * 16);
    pb0 = s[0]; pb1 = s[1];
  };
  auto stA = [&](int buf) { *(uint4*)&sA[buf][am * SA1 + ap * 8] = pa; };
  auto stB = [&](int buf) {
    uint4* d = (uint4*)&sB[buf][bn * SB1 + bp * 16];
    d[0] = pb0; d[1] = pb1;
  };
  gA(0); gB(0); stA(0); stB(0);
#endif
  __syncthreads();

  v8f z = {0, 0, 0, 0, 0, 0, 0, 0};
  v8f acc[2][2] = {{z, z}, {z, z}};

  const int NK = DD / 32;                      // 16 k-chunks
  for (int kk = 0; kk < NK; ++kk) {
    const int cur = kk & 1;
    const bool more = (kk + 1 < NK);
#if HAVE_ASYNC_LDS
    if (more) stage(cur ^ 1, (kk + 1) * 32);
#else
    if (more) { gA((kk + 1) * 32); gB((kk + 1) * 32); }
#endif

    Frag a[2];
#pragma unroll
    for (int mi = 0; mi < 2; ++mi) {
      const int m = 32 * wm + 16 * mi + ln;
      a[mi].q[0] = *(const uint4*)&sA[cur][m * SA1 + 8 * h];
      a[mi].q[1] = *(const uint4*)&sA[cur][m * SA1 + 16 + 8 * h];
    }
#pragma unroll
    for (int ni = 0; ni < 2; ++ni) {
      Frag bfr;
      const int n = 32 * wn + 16 * ni + ln;
      bfr.q[0] = *(const uint4*)&sB[cur][n * SB1 + 8 * h];
      bfr.q[1] = *(const uint4*)&sB[cur][n * SB1 + 16 + 8 * h];
      acc[0][ni] = wmma_bf16(a[0], bfr, acc[0][ni]);
      acc[1][ni] = wmma_bf16(a[1], bfr, acc[1][ni]);
    }
#if HAVE_ASYNC_LDS
    if (more) wait_async0();                  // prefetch overlapped with WMMAs above
#else
    if (more) { stA(cur ^ 1); stB(cur ^ 1); }
#endif
    __syncthreads();
  }

#pragma unroll
  for (int mi = 0; mi < 2; ++mi)
#pragma unroll
    for (int ni = 0; ni < 2; ++ni) {
      const int j = n0 + 32 * wn + 16 * ni + ln;
      const float bj = bias[j];
#pragma unroll
      for (int e = 0; e < 8; ++e) {
        const int m = m0 + 32 * wm + 16 * mi + e + 8 * h;
        P[(int64_t)m * JJ + j] = acc[mi][ni][e] + bj;
      }
    }
}

// ---------------------------------------------------------------- stage 2: fused joint
// out[r][V] = tanh(encP[b,t] + decP[b,u]) @ WoT^T + b_out, A tile resident in LDS.
#define F_MT 64
#define F_NT 256
#define F_SA 648   // padded row stride (bank-conflict-free b128 A loads)
#define F_SB 40

__global__ __launch_bounds__(256) void joint_fused(const float* __restrict__ encP,
                                                   const float* __restrict__ decP,
                                                   const uint16_t* __restrict__ WoT,
                                                   const float* __restrict__ bout,
                                                   float* __restrict__ out) {
  __shared__ uint16_t sA[F_MT * F_SA];        // 82,944 B, resident joint tile (bf16)
  __shared__ uint16_t sB[2][F_NT * F_SB];     // 2 x 20,480 B, W_out^T k-chunks
  const int tid = threadIdx.x;
  const int64_t r0 = (int64_t)blockIdx.x * F_MT;

  // ---- phase 1: build A tile, tanh computed exactly once per element
  for (int i = tid; i < F_MT * (JJ / 2); i += 256) {
    const int rr = i / (JJ / 2);
    const int jp = (i - rr * (JJ / 2)) * 2;
    const int64_t r = r0 + rr;
    const int u = (int)(r & (UU - 1));
    const int t = (int)((r >> 6) & (TT - 1));
    const int b = (int)(r >> 14);
    const float2 e2 = *(const float2*)(encP + (int64_t)(b * TT + t) * JJ + jp);
    const float2 d2 = *(const float2*)(decP + (int64_t)(b * UU + u) * JJ + jp);
    *(uint32_t*)&sA[rr * F_SA + jp] = pack2(fast_tanh(e2.x + d2.x), fast_tanh(e2.y + d2.y));
  }
  __syncthreads();

  const int lane = tid & 31, w = tid >> 5;
  const int wm = w & 1, wn = w >> 1;          // 2 M-waves x 4 N-waves -> 32x64 per wave
  const int ln = lane & 15, h = lane >> 4;

#if HAVE_ASYNC_LDS
  auto stB_async = [&](int buf, int ncol0_, int k) {  // one 64B row per thread, DMA->LDS
    const uint16_t* g = WoT + (int64_t)(ncol0_ + tid) * JJ + k;
    uint16_t* l = &sB[buf][tid * F_SB];
    __builtin_amdgcn_global_load_async_to_lds_b128(AS1CAST(g), AS3CAST(l), 0, 0);
    __builtin_amdgcn_global_load_async_to_lds_b128(AS1CAST(g), AS3CAST(l), 16, 0);
    __builtin_amdgcn_global_load_async_to_lds_b128(AS1CAST(g), AS3CAST(l), 32, 0);
    __builtin_amdgcn_global_load_async_to_lds_b128(AS1CAST(g), AS3CAST(l), 48, 0);
  };
#else
  uint4 p[4];
  auto gB = [&](int ncol0_, int k) {
    const uint4* s = (const uint4*)(WoT + (int64_t)(ncol0_ + tid) * JJ + k);
    p[0] = s[0]; p[1] = s[1]; p[2] = s[2]; p[3] = s[3];
  };
  auto stB = [&](int buf) {
    uint4* d = (uint4*)&sB[buf][tid * F_SB];
    d[0] = p[0]; d[1] = p[1]; d[2] = p[2]; d[3] = p[3];
  };
#endif

  for (int pass = 0; pass < VV / F_NT; ++pass) {  // 4 passes of 256 columns
    const int ncol0 = pass * F_NT;
#if HAVE_ASYNC_LDS
    stB_async(0, ncol0, 0);
    wait_async0();
#else
    gB(ncol0, 0); stB(0);
#endif
    __syncthreads();

    v8f z = {0, 0, 0, 0, 0, 0, 0, 0};
    v8f acc[2][4];
#pragma unroll
    for (int mi = 0; mi < 2; ++mi)
#pragma unroll
      for (int ni = 0; ni < 4; ++ni) acc[mi][ni] = z;

    const int NK = JJ / 32;                   // 20 k-chunks
    for (int kk = 0; kk < NK; ++kk) {
      const int cur = kk & 1;
      const bool more = (kk + 1 < NK);
#if HAVE_ASYNC_LDS
      if (more) stB_async(cur ^ 1, ncol0, (kk + 1) * 32);  // DMA next chunk
#else
      if (more) gB(ncol0, (kk + 1) * 32);     // prefetch into regs
#endif

      const int k = kk * 32;
      Frag a[2];
#pragma unroll
      for (int mi = 0; mi < 2; ++mi) {
        const int m = 32 * wm + 16 * mi + ln;
        a[mi].q[0] = *(const uint4*)&sA[m * F_SA + k + 8 * h];
        a[mi].q[1] = *(const uint4*)&sA[m * F_SA + k + 16 + 8 * h];
      }
#pragma unroll
      for (int ni = 0; ni < 4; ++ni) {
        Frag bfr;
        const int n = 64 * wn + 16 * ni + ln;
        bfr.q[0] = *(const uint4*)&sB[cur][n * F_SB + 8 * h];
        bfr.q[1] = *(const uint4*)&sB[cur][n * F_SB + 16 + 8 * h];
        acc[0][ni] = wmma_bf16(a[0], bfr, acc[0][ni]);
        acc[1][ni] = wmma_bf16(a[1], bfr, acc[1][ni]);
      }
#if HAVE_ASYNC_LDS
      if (more) wait_async0();                // async fill overlapped with WMMAs
#else
      if (more) stB(cur ^ 1);                 // fill other buffer
#endif
      __syncthreads();
    }

#pragma unroll
    for (int mi = 0; mi < 2; ++mi)
#pragma unroll
      for (int ni = 0; ni < 4; ++ni) {
        const int col = ncol0 + 64 * wn + 16 * ni + ln;
        const float bc = bout[col];
#pragma unroll
        for (int e = 0; e < 8; ++e) {
          const int64_t r = r0 + 32 * wm + 16 * mi + e + 8 * h;
          out[r * VV + col] = acc[mi][ni][e] + bc;
        }
      }
  }
}

// ---------------------------------------------------------------- launch
extern "C" void kernel_launch(void* const* d_in, const int* in_sizes, int n_in,
                              void* d_out, int out_size, void* d_ws, size_t ws_size,
                              hipStream_t stream) {
  const float* enc_out = (const float*)d_in[0];  // (8,256,1,512)
  const float* dec_out = (const float*)d_in[1];  // (8,1,64,512)
  const float* W_enc   = (const float*)d_in[2];  // (512,640)
  const float* b_enc   = (const float*)d_in[3];  // (640)
  const float* W_dec   = (const float*)d_in[4];  // (512,640)
  const float* b_dec   = (const float*)d_in[5];  // (640)
  const float* W_out   = (const float*)d_in[6];  // (640,1024)
  const float* b_out   = (const float*)d_in[7];  // (1024)
  float* out = (float*)d_out;                    // (8,256,64,1024)

  char* ws = (char*)d_ws;                        // ~11.8 MB total
  float*    encP = (float*)ws;     ws += (size_t)2048 * 640 * 4;  // enc_proj f32
  float*    decP = (float*)ws;     ws += (size_t)512  * 640 * 4;  // dec_proj f32
  uint16_t* WoT  = (uint16_t*)ws;  ws += (size_t)1024 * 640 * 2;  // W_out^T bf16
  uint16_t* WeT  = (uint16_t*)ws;  ws += (size_t)640  * 512 * 2;  // W_enc^T bf16
  uint16_t* WdT  = (uint16_t*)ws;  ws += (size_t)640  * 512 * 2;  // W_dec^T bf16
  uint16_t* encB = (uint16_t*)ws;  ws += (size_t)2048 * 512 * 2;  // enc_out bf16
  uint16_t* decB = (uint16_t*)ws;  ws += (size_t)512  * 512 * 2;  // dec_out bf16

  // stage 0: precision conversion / weight transposition
  cvt_pairs<<<(2048 * 512 / 2 + 255) / 256, 256, 0, stream>>>(enc_out, (uint32_t*)encB, 2048 * 512 / 2);
  cvt_pairs<<<(512 * 512 / 2 + 255) / 256, 256, 0, stream>>>(dec_out, (uint32_t*)decB, 512 * 512 / 2);
  cvtT<<<(512 * 640 + 255) / 256, 256, 0, stream>>>(W_enc, WeT, 512, 640);
  cvtT<<<(512 * 640 + 255) / 256, 256, 0, stream>>>(W_dec, WdT, 512, 640);
  cvtT<<<(640 * 1024 + 255) / 256, 256, 0, stream>>>(W_out, WoT, 640, 1024);

  // stage 1: projections (+bias), f32 results
  proj_gemm<<<dim3(2048 / P_MT, JJ / P_NT), 256, 0, stream>>>(encB, WeT, b_enc, encP);
  proj_gemm<<<dim3(512 / P_MT, JJ / P_NT), 256, 0, stream>>>(decB, WdT, b_dec, decP);

  // stage 2: fused tanh(enc+dec) @ W_out + b_out
  joint_fused<<<(BB * TT * UU) / F_MT, 256, 0, stream>>>(encP, decP, WoT, b_out, out);
}
